// TopoDP_66563403154019
// MI455X (gfx1250) — matrix operand order repaired and checked by reading, not compiled, not run on hardware
//
#include <hip/hip_runtime.h>
#include <hip/hip_bf16.h>
#include <stdint.h>

// ---------- CDNA5 WMMA types ----------
typedef __attribute__((ext_vector_type(16))) __bf16 bf16x16;
typedef __attribute__((ext_vector_type(8)))  float  f32x8;

union ABf { bf16x16 v; uint4 u[2]; };

__device__ __forceinline__ unsigned short f2bf(float f) {
  uint32_t u = __float_as_uint(f);
  uint32_t r = u + 0x7FFFu + ((u >> 16) & 1u);          // round-to-nearest-even
  if ((u & 0x7FFFFFFFu) > 0x7F800000u) r = u | 0x00400000u;  // quiet NaN
  return (unsigned short)(r >> 16);
}

// ---------- prep: fp32 -> bf16 ----------
__global__ void k_conv_x(const float* __restrict__ x, unsigned short* __restrict__ xbf, int n) {
  int i = blockIdx.x * blockDim.x + threadIdx.x;
  if (i < n) xbf[i] = f2bf(x[i]);
}

// W0 [R,128,64] -> w0t [R,64,128] bf16 (column-major per relation => contiguous K for B tiles)
__global__ void k_w0t(const float* __restrict__ W, unsigned short* __restrict__ Wt, int R) {
  int i = blockIdx.x * blockDim.x + threadIdx.x;
  int tot = R * 64 * 128;
  if (i >= tot) return;
  int r = i / (64 * 128), rem = i % (64 * 128);
  int n = rem / 128, k = rem % 128;
  Wt[i] = f2bf(W[((size_t)r * 128 + k) * 64 + n]);
}

// W [R,64,16] -> Wt [R,16,64] bf16
__global__ void k_w16t(const float* __restrict__ W, unsigned short* __restrict__ Wt, int R) {
  int i = blockIdx.x * blockDim.x + threadIdx.x;
  int tot = R * 16 * 64;
  if (i >= tot) return;
  int r = i / (16 * 64), rem = i % (16 * 64);
  int n = rem / 64, k = rem % 64;
  Wt[i] = f2bf(W[((size_t)r * 64 + k) * 16 + n]);
}

// bias sums over relations (hetero aggregate='sum')
__global__ void k_bias(const float* __restrict__ b0, const float* __restrict__ bmu,
                       const float* __restrict__ bls, float* __restrict__ s0,
                       float* __restrict__ smu, float* __restrict__ sls, int R) {
  int t = threadIdx.x;
  if (t < 64) {
    float a = 0.f;
    for (int r = 0; r < R; ++r) a += b0[r * 64 + t];
    s0[t] = a;
  } else if (t < 80) {
    int c = t - 64; float a = 0.f;
    for (int r = 0; r < R; ++r) a += bmu[r * 16 + c];
    smu[c] = a;
  } else if (t < 96) {
    int c = t - 80; float a = 0.f;
    for (int r = 0; r < R; ++r) a += bls[r * 16 + c];
    sls[c] = a;
  }
}

// ---------- degrees ----------
__global__ void k_degrees(const int* __restrict__ src, const int* __restrict__ dst,
                          float* __restrict__ dout, float* __restrict__ din, int E, int N) {
  int e = blockIdx.x * blockDim.x + threadIdx.x;
  int r = blockIdx.y;
  if (e >= E) return;
  atomicAdd(dout + (size_t)r * N + src[(size_t)r * E + e], 1.0f);
  atomicAdd(din  + (size_t)r * N + dst[(size_t)r * E + e], 1.0f);
}

__global__ void k_rsqrt(float* __restrict__ v, int n) {
  int i = blockIdx.x * blockDim.x + threadIdx.x;
  if (i < n) v[i] = rsqrtf(fmaxf(v[i], 1.0f));
}

// ---------- WMMA GEMM: P = (x @ W0[r]) * s_out[r][:,None] ----------
// one wave computes a 16-row x 64-col tile; K=128 in 4 steps of 32 (bf16 WMMA)
__global__ void __launch_bounds__(256)
k_gemm1(const unsigned short* __restrict__ xbf,      // [N,128] bf16
        const unsigned short* __restrict__ w0t,      // [64,128] bf16 (one relation)
        const float* __restrict__ sout,              // [N]
        float* __restrict__ P, int N) {
  int lane = threadIdx.x & 31;
  int tile = blockIdx.x * (blockDim.x >> 5) + (threadIdx.x >> 5);
  int ntiles = (N + 15) >> 4;
  if (tile >= ntiles) return;
  int row0 = tile << 4;
  int m = lane & 15, hi = lane >> 4;
  int ar = min(row0 + m, N - 1);
  const unsigned short* ap = xbf + (size_t)ar * 128;
  f32x8 acc0 = {}, acc1 = {}, acc2 = {}, acc3 = {};
#pragma unroll
  for (int k0 = 0; k0 < 128; k0 += 32) {
    ABf a;
    a.u[0] = *(const uint4*)(ap + k0 + hi * 16);      // K = kb..kb+7
    a.u[1] = *(const uint4*)(ap + k0 + hi * 16 + 8);  // K = kb+8..kb+15
    const unsigned short* bp = w0t + k0 + hi * 16;    // per-lane contiguous 16 K-values
    ABf b0, b1, b2, b3;
    b0.v = *(const bf16x16*)(bp + (size_t)(0 * 16 + m) * 128);
    b1.v = *(const bf16x16*)(bp + (size_t)(1 * 16 + m) * 128);
    b2.v = *(const bf16x16*)(bp + (size_t)(2 * 16 + m) * 128);
    b3.v = *(const bf16x16*)(bp + (size_t)(3 * 16 + m) * 128);
    acc0 = __builtin_amdgcn_wmma_f32_16x16x32_bf16(false, a.v, false, b0.v, (short)0, acc0, false, false);
    acc1 = __builtin_amdgcn_wmma_f32_16x16x32_bf16(false, a.v, false, b1.v, (short)0, acc1, false, false);
    acc2 = __builtin_amdgcn_wmma_f32_16x16x32_bf16(false, a.v, false, b2.v, (short)0, acc2, false, false);
    acc3 = __builtin_amdgcn_wmma_f32_16x16x32_bf16(false, a.v, false, b3.v, (short)0, acc3, false, false);
  }
#pragma unroll
  for (int v = 0; v < 8; ++v) {
    int mr = row0 + v + hi * 8;                       // C layout: M = v + 8*hi, N = m
    if (mr < N) {
      float s = sout[mr];
      float* op = P + (size_t)mr * 64 + m;
      op[0]  = acc0[v] * s;
      op[16] = acc1[v] * s;
      op[32] = acc2[v] * s;
      op[48] = acc3[v] * s;
    }
  }
}

// ---------- scatter layer1: h1acc[dst] += P[src] * s_in[r][dst] ----------
__global__ void k_scatter1(const float* __restrict__ P, const int* __restrict__ src,
                           const int* __restrict__ dst, const float* __restrict__ sin_,
                           float* __restrict__ acc, int E) {
  int idx = blockIdx.x * blockDim.x + threadIdx.x;   // E*16 threads, 4 floats each
  if (idx >= E * 16) return;
  int e = idx >> 4, c = (idx & 15) << 2;
  int s = src[e], d = dst[e];
  float4 v = *(const float4*)(P + (size_t)s * 64 + c);
  float sc = sin_[d];
  float* o = acc + (size_t)d * 64 + c;
  atomicAdd(o + 0, v.x * sc);
  atomicAdd(o + 1, v.y * sc);
  atomicAdd(o + 2, v.z * sc);
  atomicAdd(o + 3, v.w * sc);
}

// ---------- relu epilogue + bf16 for layer 2 ----------
__global__ void k_epi1(const float* __restrict__ acc, const float* __restrict__ bsum,
                       unsigned short* __restrict__ h1bf, int n) {
  int i = blockIdx.x * blockDim.x + threadIdx.x;
  if (i >= n) return;
  float v = acc[i] + bsum[i & 63];
  h1bf[i] = f2bf(fmaxf(v, 0.0f));
}

// ---------- fused mu/ls WMMA GEMM: K=64, N=16, shared A tile ----------
__global__ void __launch_bounds__(256)
k_gemm2(const unsigned short* __restrict__ h1bf,     // [N,64] bf16
        const unsigned short* __restrict__ wmut,     // [16,64] bf16
        const unsigned short* __restrict__ wlst,     // [16,64] bf16
        const float* __restrict__ sout,
        float* __restrict__ Pmu, float* __restrict__ Pls, int N) {
  int lane = threadIdx.x & 31;
  int tile = blockIdx.x * (blockDim.x >> 5) + (threadIdx.x >> 5);
  int ntiles = (N + 15) >> 4;
  if (tile >= ntiles) return;
  int row0 = tile << 4;
  int m = lane & 15, hi = lane >> 4;
  int ar = min(row0 + m, N - 1);
  const unsigned short* ap = h1bf + (size_t)ar * 64;
  f32x8 am = {}, al = {};
#pragma unroll
  for (int k0 = 0; k0 < 64; k0 += 32) {
    ABf a;
    a.u[0] = *(const uint4*)(ap + k0 + hi * 16);
    a.u[1] = *(const uint4*)(ap + k0 + hi * 16 + 8);
    ABf bm, bl;
    bm.v = *(const bf16x16*)(wmut + (size_t)m * 64 + k0 + hi * 16);
    bl.v = *(const bf16x16*)(wlst + (size_t)m * 64 + k0 + hi * 16);
    am = __builtin_amdgcn_wmma_f32_16x16x32_bf16(false, a.v, false, bm.v, (short)0, am, false, false);
    al = __builtin_amdgcn_wmma_f32_16x16x32_bf16(false, a.v, false, bl.v, (short)0, al, false, false);
  }
#pragma unroll
  for (int v = 0; v < 8; ++v) {
    int mr = row0 + v + hi * 8;
    if (mr < N) {
      float s = sout[mr];
      Pmu[(size_t)mr * 16 + m] = am[v] * s;
      Pls[(size_t)mr * 16 + m] = al[v] * s;
    }
  }
}

// ---------- scatter layer2: mean/ls accumulators, fused ----------
__global__ void k_scatter2(const float* __restrict__ Pmu, const float* __restrict__ Pls,
                           const int* __restrict__ src, const int* __restrict__ dst,
                           const float* __restrict__ sin_, float* __restrict__ mAcc,
                           float* __restrict__ lAcc, int E) {
  int idx = blockIdx.x * blockDim.x + threadIdx.x;   // E*4 threads, 4 floats each
  if (idx >= E * 4) return;
  int e = idx >> 2, c = (idx & 3) << 2;
  int s = src[e], d = dst[e];
  float sc = sin_[d];
  float4 vm = *(const float4*)(Pmu + (size_t)s * 16 + c);
  float4 vl = *(const float4*)(Pls + (size_t)s * 16 + c);
  float* om = mAcc + (size_t)d * 16 + c;
  float* ol = lAcc + (size_t)d * 16 + c;
  atomicAdd(om + 0, vm.x * sc); atomicAdd(om + 1, vm.y * sc);
  atomicAdd(om + 2, vm.z * sc); atomicAdd(om + 3, vm.w * sc);
  atomicAdd(ol + 0, vl.x * sc); atomicAdd(ol + 1, vl.y * sc);
  atomicAdd(ol + 2, vl.z * sc); atomicAdd(ol + 3, vl.w * sc);
}

// ---------- reparameterize: h = mean + N(0,1)*exp(log_std) ----------
__device__ __forceinline__ uint32_t pcg(uint32_t i) {
  uint32_t s = i * 747796405u + 2891336453u;
  uint32_t w = ((s >> ((s >> 28) + 4u)) ^ s) * 277803737u;
  return (w >> 22) ^ w;
}
__device__ __forceinline__ float gauss(uint32_t i) {
  float u1 = (pcg(i) + 0.5f) * 2.3283064365386963e-10f;
  float u2 = (pcg(i ^ 0x9E3779B9u) + 0.5f) * 2.3283064365386963e-10f;
  return sqrtf(-2.0f * logf(u1)) * cosf(6.28318530717959f * u2);
}

__global__ void k_final(const float* __restrict__ mAcc, const float* __restrict__ lAcc,
                        const float* __restrict__ bmu, const float* __restrict__ bls,
                        float* __restrict__ hout, int n) {
  int i = blockIdx.x * blockDim.x + threadIdx.x;
  if (i >= n) return;
  int c = i & 15;
  float mv = mAcc[i] + bmu[c];
  float lv = lAcc[i] + bls[c];
  hout[i] = mv + gauss((uint32_t)i) * expf(lv);
}

// ---------- edge dot-product scores ----------
__global__ void k_scores(const int* __restrict__ ps, const int* __restrict__ pd,
                         const int* __restrict__ ns, const int* __restrict__ nd,
                         const float* __restrict__ h, float* __restrict__ out, int EP) {
  int i = blockIdx.x * blockDim.x + threadIdx.x;
  if (i >= 2 * EP) return;
  int s, d;
  if (i < EP) { s = ps[i]; d = pd[i]; }
  else        { int j = i - EP; s = ns[j]; d = nd[j]; }
  const float4* a = (const float4*)(h + (size_t)s * 16);
  const float4* b = (const float4*)(h + (size_t)d * 16);
  float acc = 0.f;
#pragma unroll
  for (int k = 0; k < 4; ++k) {
    float4 x = a[k], y = b[k];
    acc += x.x * y.x + x.y * y.y + x.z * y.z + x.w * y.w;
  }
  out[i] = acc;
}

extern "C" void kernel_launch(void* const* d_in, const int* in_sizes, int n_in,
                              void* d_out, int out_size, void* d_ws, size_t ws_size,
                              hipStream_t stream) {
  const float* x    = (const float*)d_in[0];
  const float* W0   = (const float*)d_in[1];
  const float* b0   = (const float*)d_in[2];
  const float* Wmu  = (const float*)d_in[3];
  const float* bmu  = (const float*)d_in[4];
  const float* Wls  = (const float*)d_in[5];
  const float* bls  = (const float*)d_in[6];
  const int* esrc   = (const int*)d_in[7];
  const int* edst   = (const int*)d_in[8];
  const int* psrc   = (const int*)d_in[9];
  const int* pdst   = (const int*)d_in[10];
  const int* nsrc   = (const int*)d_in[11];
  const int* ndst   = (const int*)d_in[12];
  float* out        = (float*)d_out;

  const int N  = in_sizes[0] / 128;
  const int R  = in_sizes[1] / (128 * 64);
  const int E  = in_sizes[7] / R;
  const int EP = in_sizes[9];

  // ---- workspace layout (256B aligned) ----
  size_t off = 0;
  auto alloc = [&](size_t bytes) -> void* {
    void* p = (char*)d_ws + off;
    off += (bytes + 255) & ~(size_t)255;
    return p;
  };
  // zeroed region first (single memset)
  float* s_out  = (float*)alloc((size_t)R * N * 4);
  float* s_in   = (float*)alloc((size_t)R * N * 4);
  float* h1acc  = (float*)alloc((size_t)N * 64 * 4);
  float* mAcc   = (float*)alloc((size_t)N * 16 * 4);
  float* lAcc   = (float*)alloc((size_t)N * 16 * 4);
  size_t zbytes = off;
  unsigned short* xbf  = (unsigned short*)alloc((size_t)N * 128 * 2);
  unsigned short* h1bf = (unsigned short*)alloc((size_t)N * 64 * 2);
  float* P    = (float*)alloc((size_t)N * 64 * 4);
  float* Pmu  = (float*)alloc((size_t)N * 16 * 4);
  float* Pls  = (float*)alloc((size_t)N * 16 * 4);
  unsigned short* w0t  = (unsigned short*)alloc((size_t)R * 64 * 128 * 2);
  unsigned short* wmut = (unsigned short*)alloc((size_t)R * 16 * 64 * 2);
  unsigned short* wlst = (unsigned short*)alloc((size_t)R * 16 * 64 * 2);
  float* bsum0 = (float*)alloc(64 * 4);
  float* bsmu  = (float*)alloc(16 * 4);
  float* bsls  = (float*)alloc(16 * 4);
  (void)ws_size; (void)n_in; (void)out_size;

  hipMemsetAsync(d_ws, 0, zbytes, stream);

  // prep: bf16 conversions + transposed weights + bias sums
  k_conv_x<<<(N * 128 + 255) / 256, 256, 0, stream>>>(x, xbf, N * 128);
  k_w0t<<<(R * 64 * 128 + 255) / 256, 256, 0, stream>>>(W0, w0t, R);
  k_w16t<<<(R * 16 * 64 + 255) / 256, 256, 0, stream>>>(Wmu, wmut, R);
  k_w16t<<<(R * 16 * 64 + 255) / 256, 256, 0, stream>>>(Wls, wlst, R);
  k_bias<<<1, 96, 0, stream>>>(b0, bmu, bls, bsum0, bsmu, bsls, R);

  // degrees -> rsqrt(clip(deg,1))
  dim3 gdeg((E + 255) / 256, R);
  k_degrees<<<gdeg, 256, 0, stream>>>(esrc, edst, s_out, s_in, E, N);
  k_rsqrt<<<(2 * R * N + 255) / 256, 256, 0, stream>>>(s_out, 2 * R * N);

  const int ntiles = (N + 15) / 16;
  const int gemmBlocks = (ntiles + 7) / 8;   // 8 waves/block, 1 tile/wave

  // layer 1 per relation: WMMA project -> atomic scatter
  for (int r = 0; r < R; ++r) {
    k_gemm1<<<gemmBlocks, 256, 0, stream>>>(xbf, w0t + (size_t)r * 64 * 128,
                                            s_out + (size_t)r * N, P, N);
    k_scatter1<<<(E * 16 + 255) / 256, 256, 0, stream>>>(
        P, esrc + (size_t)r * E, edst + (size_t)r * E, s_in + (size_t)r * N, h1acc, E);
  }
  k_epi1<<<(N * 64 + 255) / 256, 256, 0, stream>>>(h1acc, bsum0, h1bf, N * 64);

  // layers 2+3 per relation: fused mu/ls WMMA -> fused scatter
  for (int r = 0; r < R; ++r) {
    k_gemm2<<<gemmBlocks, 256, 0, stream>>>(h1bf, wmut + (size_t)r * 16 * 64,
                                            wlst + (size_t)r * 16 * 64,
                                            s_out + (size_t)r * N, Pmu, Pls, N);
    k_scatter2<<<(E * 4 + 255) / 256, 256, 0, stream>>>(
        Pmu, Pls, esrc + (size_t)r * E, edst + (size_t)r * E, s_in + (size_t)r * N,
        mAcc, lAcc, E);
  }

  // reparameterize -> h goes straight into d_out[2*EP ..]
  float* hout = out + (size_t)2 * EP;
  k_final<<<(N * 16 + 255) / 256, 256, 0, stream>>>(mAcc, lAcc, bsmu, bsls, hout, N * 16);

  // pos/neg scores into d_out[0..2*EP)
  k_scores<<<(2 * EP + 255) / 256, 256, 0, stream>>>(psrc, pdst, nsrc, ndst, hout, out, EP);
}
